// RecurrentCorrector_43877385896057
// MI455X (gfx1250) — matrix operand order repaired and checked by reading, not compiled, not run on hardware
//
// MI455X (gfx1250, wave32) RecurrentCorrector.
// Strategy: hoist all non-recurrent GEMMs (input gate projections, proj_keys)
// into large bf16 WMMA GEMMs; per-timestep recurrence uses small WMMA GEMMs
// whose bf16 weights (~51MB) live in the 192MB L2. All matmuls lower to
// v_wmma_f32_16x16x32_bf16. The GEMM k-loop is LDS-staged with a barrier-paced
// double buffer: global loads for stage n+1 are in flight across the entire
// stage n (ds_load fragments + 4 WMMAs), instead of stalling each WMMA on
// s_wait_loadcnt 0. LDS rows padded to 80B for conflict-free ds_load_b128.
#include <hip/hip_runtime.h>
#include <math.h>

typedef __bf16 bf16;
typedef __attribute__((ext_vector_type(16))) __bf16 v16bf;
typedef __attribute__((ext_vector_type(8)))  __bf16 v8bf;
typedef __attribute__((ext_vector_type(8)))  float  v8f;

union AF { v16bf v; v8bf h[2]; };

#define B_   64
#define T_   128
#define S_   128
#define E_   512
#define H_   1024
#define D_   1024
#define K_   1024
#define G3_  (3 * H_)           // 3072
#define IN2_ (2 * D_ + 2 * H_)  // 4096
#define BT_  (B_ * T_)          // 8192
#define PAD_ 40                 // LDS row stride in bf16 (32 data + 8 pad = 80B)

// ---------------------------------------------------------------------------
// NT GEMM: C[M,N] = A[M,K] * Bw[N,K]^T  (+Cin) (+bias) (tanh) -> f32 / bf16
// Block: 256 threads = 8 waves; tile 64(M) x 128(N); wave tile 16x64.
// Requires: M % 64 == 0, N % 128 == 0, K % 32 == 0 (all call sites comply).
// Double-buffered LDS staging: per 32-wide k-stage each thread fetches
// 3 x b128 from global (2 B chunks + 1 A chunk) for stage n+1 while the
// waves consume stage n from LDS (10 x ds_load_b128 + 4 WMMA per wave).
// ---------------------------------------------------------------------------
__global__ __launch_bounds__(256) void gemm_bf16_wmma(
    const bf16* __restrict__ A, int lda,
    const bf16* __restrict__ Bw, int ldb,
    const float* __restrict__ Cin, int ldci,
    float* __restrict__ Cout, int ldco,
    bf16* __restrict__ Obf, int ldob,
    const float* __restrict__ bias, int act,
    int M, int N, int K)
{
    __shared__ bf16 ldsB[2][128 * PAD_];   // 2 x 10240 B
    __shared__ bf16 ldsA[2][64 * PAD_];    // 2 x  5120 B

    const int tid  = threadIdx.x;
    const int lane = tid & 31;
    const int wid  = tid >> 5;       // 0..7
    const int wm   = wid & 3;        // M sub-tile index (0..3)
    const int wn   = wid >> 2;       // N half (0..1)
    const int hf   = lane >> 4;      // half-wave
    const int l16  = lane & 15;

    const int mBase = blockIdx.y * 64;
    const int nBase = blockIdx.x * 128;

    // Cooperative fetch mapping: B tile = 128 rows x 4 chunks(16B) = 512 chunks
    // -> 2/thread (c0 = tid, c1 = tid+256). A tile = 64 rows x 4 = 256 -> 1/thread.
    const int bRow0 = tid >> 2, bSub = tid & 3;     // c1 row = bRow0 + 64, same sub
    const int aRow  = tid >> 2, aSub = tid & 3;

    const bf16* gB0 = Bw + (size_t)(nBase + bRow0) * ldb + bSub * 8;
    const bf16* gB1 = Bw + (size_t)(nBase + bRow0 + 64) * ldb + bSub * 8;
    const bf16* gA  = A  + (size_t)(mBase + aRow) * lda + aSub * 8;

    v8bf rb0, rb1, ra;
    auto fetch = [&](int k) {
        rb0 = *(const v8bf*)(gB0 + k);
        rb1 = *(const v8bf*)(gB1 + k);
        ra  = *(const v8bf*)(gA + k);
        __builtin_prefetch(gB0 + k + 256, 0, 3);   // global_prefetch_b8
        __builtin_prefetch(gB1 + k + 256, 0, 3);
    };
    auto stage = [&](int st) {
        *(v8bf*)(&ldsB[st][bRow0 * PAD_ + bSub * 8]) = rb0;
        *(v8bf*)(&ldsB[st][(bRow0 + 64) * PAD_ + bSub * 8]) = rb1;
        *(v8bf*)(&ldsA[st][aRow * PAD_ + aSub * 8]) = ra;
    };

    v8f acc[4] = {};

    fetch(0);
    stage(0);
    __syncthreads();

    int st = 0;
    for (int k = 0; k < K; k += 32) {
        const bool more = (k + 32) < K;
        if (more) fetch(k + 32);          // global loads span the whole stage

        // A frag (16x32, half-wave K swizzle) from LDS
        AF a;
        const bf16* Ab = &ldsA[st][(wm * 16 + l16) * PAD_];
        a.h[0] = *(const v8bf*)(Ab + hf * 8);
        a.h[1] = *(const v8bf*)(Ab + 16 + hf * 8);
#pragma unroll
        for (int s = 0; s < 4; ++s) {
            // B frag (32x16): lane holds weight row nBase+wn*64+s*16+l16,
            // half-wave K swizzle (same as A)
            const bf16* Bb = &ldsB[st][(wn * 64 + s * 16 + l16) * PAD_];
            AF b;
            b.h[0] = *(const v8bf*)(Bb + hf * 8);
            b.h[1] = *(const v8bf*)(Bb + 16 + hf * 8);
            acc[s] = __builtin_amdgcn_wmma_f32_16x16x32_bf16(
                false, a.v, false, b.v, (short)0, acc[s], false, false);
        }

        if (more) stage(st ^ 1);          // write alternate buffer
        __syncthreads();                  // one barrier per stage suffices
        st ^= 1;
    }

    // Epilogue: C/D layout — VGPR r: lanes0-15 -> M=r, lanes16-31 -> M=r+8
#pragma unroll
    for (int s = 0; s < 4; ++s) {
        const int col = nBase + wn * 64 + s * 16 + l16;
        const float bv = bias ? bias[col] : 0.f;
#pragma unroll
        for (int r = 0; r < 8; ++r) {
            const int row = mBase + wm * 16 + hf * 8 + r;
            float v = acc[s][r] + bv;
            if (Cin)  v += Cin[(size_t)row * ldci + col];
            if (act)  v = tanhf(v);
            if (Cout) Cout[(size_t)row * ldco + col] = v;
            if (Obf)  Obf[(size_t)row * ldob + col] = (bf16)v;
        }
    }
}

// ---------------------------------------------------------------------------
__global__ void f32_to_bf16_k(const float* __restrict__ in, bf16* __restrict__ out, int n) {
    int i = blockIdx.x * 256 + threadIdx.x;
    if (i < n) out[i] = (bf16)in[i];
}
__global__ void fill_f32_zero(float* p, int n) {
    int i = blockIdx.x * 256 + threadIdx.x;
    if (i < n) p[i] = 0.f;
}
__global__ void fill_bf16_zero(bf16* p, int n) {
    int i = blockIdx.x * 256 + threadIdx.x;
    if (i < n) p[i] = (bf16)0.f;
}

// Fused GRU gate nonlinearity + state update (+ bf16 copies for next GEMM).
__global__ void gru_elem(const float* __restrict__ gi, int gis,
                         const float* __restrict__ gh, int ghs,
                         const float* __restrict__ bih, const float* __restrict__ bhh,
                         const float* __restrict__ h_in,
                         float* __restrict__ h_out,
                         bf16* __restrict__ h_bf, int hbs,
                         bf16* __restrict__ extra_bf, int ebs,
                         int Bn, int Hn)
{
    int idx = blockIdx.x * 256 + threadIdx.x;
    if (idx >= Bn * Hn) return;
    int b = idx / Hn, j = idx - b * Hn;
    const float* gib = gi + (size_t)b * gis;
    const float* ghb = gh + (size_t)b * ghs;
    float r = 1.f / (1.f + expf(-(gib[j] + bih[j] + ghb[j] + bhh[j])));
    float z = 1.f / (1.f + expf(-(gib[Hn + j] + bih[Hn + j] + ghb[Hn + j] + bhh[Hn + j])));
    float nn = tanhf(gib[2 * Hn + j] + bih[2 * Hn + j] + r * (ghb[2 * Hn + j] + bhh[2 * Hn + j]));
    float hp = h_in ? h_in[idx] : 0.f;
    float h = (1.f - z) * nn + z * hp;
    h_out[idx] = h;
    if (h_bf)     h_bf[(size_t)b * hbs + j] = (bf16)h;
    if (extra_bf) extra_bf[(size_t)b * ebs + j] = (bf16)h;
}

// reward = h @ rW^T + rb (N=2); argmax; scale corr by (1-pred); emit outputs.
__global__ __launch_bounds__(128) void reward_scale_step(
    const float* __restrict__ h, const float* __restrict__ rW,
    const float* __restrict__ rb,
    float* __restrict__ rew_out, int rew_stride,
    float* __restrict__ corr, bf16* __restrict__ corr_bf,
    float* __restrict__ corrs_out, int corrs_stride,
    int Hn, int Dn)
{
    __shared__ float red[128];
    __shared__ float scale_s;
    int b = blockIdx.x, tid = threadIdx.x;
    int j = tid >> 6, l = tid & 63;
    const float* hb = h + (size_t)b * Hn;
    const float* wj = rW + (size_t)j * Hn;
    float p = 0.f;
    for (int k = l; k < Hn; k += 64) p += hb[k] * wj[k];
    red[tid] = p;
    __syncthreads();
    for (int st = 32; st > 0; st >>= 1) {
        if (l < st) red[tid] += red[tid + st];
        __syncthreads();
    }
    if (tid == 0) {
        float r0 = red[0] + rb[0];
        float r1 = red[64] + rb[1];
        rew_out[(size_t)b * rew_stride + 0] = r0;
        rew_out[(size_t)b * rew_stride + 1] = r1;
        scale_s = (r1 > r0) ? 0.f : 1.f;   // 1 - argmax([r0,r1])
    }
    __syncthreads();
    float sc = scale_s;
    for (int d = tid; d < Dn; d += 128) {
        float v = corr[(size_t)b * Dn + d] * sc;
        corr[(size_t)b * Dn + d] = v;
        corr_bf[(size_t)b * Dn + d] = (bf16)v;
        corrs_out[(size_t)b * corrs_stride + d] = v;
    }
}

// energies = tanh(pq + proj_keys) . eW ; masked softmax over S ;
// context = alphas . encoder ; write attns + bf16 context into hc buffer.
__global__ __launch_bounds__(256) void attention_step(
    const float* __restrict__ pq, const float* __restrict__ pkeys,
    const float* __restrict__ eW, const unsigned char* __restrict__ smask,
    const float* __restrict__ enc,
    float* __restrict__ attn_out, int attn_stride,
    bf16* __restrict__ ctx_bf, int ctx_stride,
    int S, int Hn, int Kn)
{
    __shared__ float esh[S_];
    __shared__ float mred, sred;
    int b = blockIdx.x, tid = threadIdx.x;
    if (tid < S) {
        const float* pqb = pq + (size_t)b * Hn;
        const float* pk = pkeys + ((size_t)b * S + tid) * Hn;
        float e = 0.f;
        for (int jx = 0; jx < Hn; ++jx) e += tanhf(pqb[jx] + pk[jx]) * eW[jx];
        bool m = smask ? (smask[b * S + tid] != 0) : true;
        esh[tid] = m ? e : -__builtin_inff();
    }
    __syncthreads();
    if (tid == 0) {
        float mx = -__builtin_inff();
        for (int s = 0; s < S; ++s) mx = fmaxf(mx, esh[s]);
        float sm = 0.f;
        for (int s = 0; s < S; ++s) sm += expf(esh[s] - mx);
        mred = mx; sred = sm;
    }
    __syncthreads();
    if (tid < S) {
        float a = expf(esh[tid] - mred) / sred;
        attn_out[(size_t)b * attn_stride + tid] = a;
        esh[tid] = a;
    }
    __syncthreads();
    for (int k = tid; k < Kn; k += 256) {
        const float* eb = enc + (size_t)b * S * Kn + k;
        float c = 0.f;
        for (int s = 0; s < S; ++s) c += esh[s] * eb[(size_t)s * Kn];
        ctx_bf[(size_t)b * ctx_stride + k] = (bf16)c;
    }
}

// ---------------------------------------------------------------------------
extern "C" void kernel_launch(void* const* d_in, const int* in_sizes, int n_in,
                              void* d_out, int out_size, void* d_ws, size_t ws_size,
                              hipStream_t stream)
{
    (void)in_sizes; (void)n_in; (void)out_size; (void)ws_size;
    const float* reversed_input = (const float*)d_in[0];
    const float* y_states       = (const float*)d_in[3];
    const float* encoder        = (const float*)d_in[4];
    const unsigned char* src_mask = (const unsigned char*)d_in[5];
    const float* rnn_Wih = (const float*)d_in[6];
    const float* rnn_Whh = (const float*)d_in[7];
    const float* rnn_bih = (const float*)d_in[8];
    const float* rnn_bhh = (const float*)d_in[9];
    const float* out_Wih = (const float*)d_in[10];
    const float* out_Whh = (const float*)d_in[11];
    const float* out_bih = (const float*)d_in[12];
    const float* out_bhh = (const float*)d_in[13];
    const float* corr_W  = (const float*)d_in[14];
    const float* corr_b  = (const float*)d_in[15];
    const float* reward_W = (const float*)d_in[16];
    const float* reward_b = (const float*)d_in[17];
    const float* attv_W  = (const float*)d_in[18];
    const float* attv_b  = (const float*)d_in[19];
    const float* key_W   = (const float*)d_in[20];
    const float* query_W = (const float*)d_in[21];
    const float* energy_W = (const float*)d_in[22];

    float* corrs_out = (float*)d_out;                         // (B,T,D)
    float* rew_out   = corrs_out + (size_t)BT_ * D_;          // (B,T,2)
    float* attns_out = rew_out + (size_t)BT_ * 2;             // (B,T,S)

    // --- workspace carve (256B aligned) ---
    char* wsb = (char*)d_ws;
    size_t off = 0;
    auto alloc = [&](size_t bytes) -> void* {
        void* p = wsb + off;
        off += (bytes + 255) & ~(size_t)255;
        return p;
    };
    // bf16 weights
    bf16* wRih  = (bf16*)alloc((size_t)G3_ * E_ * 2);
    bf16* wRhh  = (bf16*)alloc((size_t)G3_ * H_ * 2);
    bf16* wOih  = (bf16*)alloc((size_t)G3_ * IN2_ * 2);
    bf16* wOhh  = (bf16*)alloc((size_t)G3_ * H_ * 2);
    bf16* wCorr = (bf16*)alloc((size_t)D_ * H_ * 2);
    bf16* wAttv = (bf16*)alloc((size_t)H_ * (H_ + K_) * 2);
    bf16* wKey  = (bf16*)alloc((size_t)H_ * K_ * 2);
    bf16* wQry  = (bf16*)alloc((size_t)H_ * H_ * 2);
    // bf16 activations
    bf16* aRev  = (bf16*)alloc((size_t)BT_ * E_ * 2);
    bf16* aY    = (bf16*)alloc((size_t)BT_ * D_ * 2);
    bf16* aEnc  = (bf16*)alloc((size_t)B_ * S_ * K_ * 2);
    bf16* rnnBf = (bf16*)alloc((size_t)BT_ * H_ * 2);
    bf16* h1bf  = (bf16*)alloc((size_t)B_ * H_ * 2);
    bf16* corrBf = (bf16*)alloc((size_t)B_ * D_ * 2);
    bf16* attBf = (bf16*)alloc((size_t)B_ * H_ * 2);
    bf16* hc    = (bf16*)alloc((size_t)B_ * 2 * H_ * 2);  // [h2 | context] per row
    // f32 buffers (giBig reused: phase1 GI_rnn, then phase2 GI_comb)
    float* giBig = (float*)alloc((size_t)BT_ * G3_ * 4);
    float* pkeys = (float*)alloc((size_t)B_ * S_ * H_ * 4);
    float* ghBuf = (float*)alloc((size_t)B_ * G3_ * 4);
    float* g1Buf = (float*)alloc((size_t)B_ * G3_ * 4);
    float* h1    = (float*)alloc((size_t)B_ * H_ * 4);
    float* h2    = (float*)alloc((size_t)B_ * H_ * 4);
    float* corrF = (float*)alloc((size_t)B_ * D_ * 4);
    float* pqBuf = (float*)alloc((size_t)B_ * H_ * 4);

    auto conv = [&](const float* in, bf16* out, size_t n) {
        f32_to_bf16_k<<<dim3((unsigned)((n + 255) / 256)), dim3(256), 0, stream>>>(in, out, (int)n);
    };
    auto gemm = [&](const bf16* A, int lda, const bf16* Bw, int ldb,
                    const float* Cin, int ldci, float* Cout, int ldco,
                    bf16* Obf, int ldob, const float* bias, int act,
                    int M, int N, int K) {
        dim3 g(N / 128, M / 64);
        gemm_bf16_wmma<<<g, dim3(256), 0, stream>>>(A, lda, Bw, ldb, Cin, ldci,
                                                    Cout, ldco, Obf, ldob, bias, act, M, N, K);
    };

    // --- convert inputs/weights to bf16 (once per call) ---
    conv(reversed_input, aRev, (size_t)BT_ * E_);
    conv(y_states, aY, (size_t)BT_ * D_);
    conv(encoder, aEnc, (size_t)B_ * S_ * K_);
    conv(rnn_Wih, wRih, (size_t)G3_ * E_);
    conv(rnn_Whh, wRhh, (size_t)G3_ * H_);
    conv(out_Wih, wOih, (size_t)G3_ * IN2_);
    conv(out_Whh, wOhh, (size_t)G3_ * H_);
    conv(corr_W, wCorr, (size_t)D_ * H_);
    conv(attv_W, wAttv, (size_t)H_ * (H_ + K_));
    conv(key_W, wKey, (size_t)H_ * K_);
    conv(query_W, wQry, (size_t)H_ * H_);

    // --- zero recurrent state ---
    fill_f32_zero<<<dim3((B_ * H_ + 255) / 256), dim3(256), 0, stream>>>(h1, B_ * H_);
    fill_f32_zero<<<dim3((B_ * H_ + 255) / 256), dim3(256), 0, stream>>>(h2, B_ * H_);
    fill_bf16_zero<<<dim3((B_ * H_ + 255) / 256), dim3(256), 0, stream>>>(h1bf, B_ * H_);
    fill_bf16_zero<<<dim3((B_ * 2 * H_ + 255) / 256), dim3(256), 0, stream>>>(hc, B_ * 2 * H_);
    fill_bf16_zero<<<dim3((B_ * D_ + 255) / 256), dim3(256), 0, stream>>>(corrBf, B_ * D_);
    fill_bf16_zero<<<dim3((B_ * H_ + 255) / 256), dim3(256), 0, stream>>>(attBf, B_ * H_);

    // --- hoisted big GEMMs ---
    // proj_keys = encoder @ key_W^T   (8192 x 1024, K=1024)
    gemm(aEnc, K_, wKey, K_, nullptr, 0, pkeys, H_, nullptr, 0, nullptr, 0, B_ * S_, H_, K_);
    // GI_rnn = reversed_input @ rnn_Wih^T  (8192 x 3072, K=512) -> giBig
    gemm(aRev, E_, wRih, E_, nullptr, 0, giBig, G3_, nullptr, 0, nullptr, 0, BT_, G3_, E_);

    // --- phase 1: GRU over T (reversed); store h into rnnBf at flipped t ---
    for (int t = 0; t < T_; ++t) {
        gemm(h1bf, H_, wRhh, H_, nullptr, 0, ghBuf, G3_, nullptr, 0, nullptr, 0, B_, G3_, H_);
        gru_elem<<<dim3((B_ * H_ + 255) / 256), dim3(256), 0, stream>>>(
            giBig + (size_t)t * G3_, T_ * G3_, ghBuf, G3_, rnn_bih, rnn_bhh,
            h1, h1, h1bf, H_, rnnBf + (size_t)(T_ - 1 - t) * H_, T_ * H_, B_, H_);
    }

    // --- hoisted phase-2 input projection (overwrites giBig) ---
    // GI_comb = y_states @ out_Wih[:, :1024]^T + rnn_outputs @ out_Wih[:, 1024:2048]^T
    gemm(aY, D_, wOih, IN2_, nullptr, 0, giBig, G3_, nullptr, 0, nullptr, 0, BT_, G3_, D_);
    gemm(rnnBf, H_, wOih + 1024, IN2_, giBig, G3_, giBig, G3_, nullptr, 0, nullptr, 0, BT_, G3_, H_);

    // --- phase 2: corrector scan ---
    for (int t = 0; t < T_; ++t) {
        // g1 = GI_comb[t] + corr_prev @ out_Wih[:,2048:3072]^T + att_prev @ out_Wih[:,3072:]^T
        gemm(corrBf, D_, wOih + 2048, IN2_, giBig + (size_t)t * G3_, T_ * G3_,
             g1Buf, G3_, nullptr, 0, nullptr, 0, B_, G3_, D_);
        gemm(attBf, H_, wOih + 3072, IN2_, g1Buf, G3_, g1Buf, G3_,
             nullptr, 0, nullptr, 0, B_, G3_, H_);
        // gh = h @ out_Whh^T   (h bf16 lives in hc[:, :H], row stride 2H)
        gemm(hc, 2 * H_, wOhh, H_, nullptr, 0, ghBuf, G3_, nullptr, 0, nullptr, 0, B_, G3_, H_);
        gru_elem<<<dim3((B_ * H_ + 255) / 256), dim3(256), 0, stream>>>(
            g1Buf, G3_, ghBuf, G3_, out_bih, out_bhh, h2, h2, hc, 2 * H_,
            (bf16*)nullptr, 0, B_, H_);
        // corr = tanh(h @ corr_W^T + corr_b)
        gemm(hc, 2 * H_, wCorr, H_, nullptr, 0, corrF, D_, nullptr, 0, corr_b, 1, B_, D_, H_);
        // reward + argmax + corr scaling + outputs
        reward_scale_step<<<dim3(B_), dim3(128), 0, stream>>>(
            h2, reward_W, reward_b, rew_out + (size_t)t * 2, T_ * 2,
            corrF, corrBf, corrs_out + (size_t)t * D_, T_ * D_, H_, D_);
        // pq = h @ query_W^T
        gemm(hc, 2 * H_, wQry, H_, nullptr, 0, pqBuf, H_, nullptr, 0, nullptr, 0, B_, H_, H_);
        // attention: energies -> softmax -> context (bf16 into hc[:, H:])
        attention_step<<<dim3(B_), dim3(256), 0, stream>>>(
            pqBuf, pkeys, energy_W, src_mask, encoder,
            attns_out + (size_t)t * S_, T_ * S_, hc + H_, 2 * H_, S_, H_, K_);
        // att_vec = tanh([h | context] @ attv_W^T + attv_b) -> attBf
        gemm(hc, 2 * H_, wAttv, H_ + K_, nullptr, 0, nullptr, 0,
             attBf, H_, attv_b, 1, B_, H_, H_ + K_);
    }
}
